// MultiHeadAttention_44487271252105
// MI455X (gfx1250) — matrix operand order repaired and checked by reading, not compiled, not run on hardware
//
#include <hip/hip_runtime.h>

typedef __attribute__((ext_vector_type(16))) _Float16 v16h;
typedef __attribute__((ext_vector_type(8)))  float    v8f;

// ---------------------------------------------------------------------------
// WMMA helper (gfx1250 wave32, 16x16x32 f16 -> f32 accumulate)
// ---------------------------------------------------------------------------
__device__ __forceinline__ v8f wmma32(v16h a, v16h b, v8f c) {
  return __builtin_amdgcn_wmma_f32_16x16x32_f16(
      /*neg_a=*/false, a, /*neg_b=*/false, b,
      /*c_mod=*/(short)0, c, /*reuse_a=*/false, /*reuse_b=*/false);
}

// ---------------------------------------------------------------------------
// Swizzled fragment-major LDS layout.
// A 16x32 f16 fragment (ISA 7.12.2): lane l=(r,hi) element e -> row=r,
//   K = (e<8 ? 8*hi+e : 16+8*hi+(e-8)).
// B 32x16 f16 fragment: lane l=(r,hi) element e -> col=r, same K as A.
// Each 16(row/col) x 32(K) subtile stored as [lane][16 halves], 20-half lane
// stride: 40B/lane -> bank = 10*l mod 64, collision-free across 32 lanes;
// fragment load = 4 consecutive b64-class LDS reads (merges to 2addr_b64).
// ---------------------------------------------------------------------------
__device__ __forceinline__ int aSwz(int row, int k32) {
  return ((row >> 4) * 640) +
         (((row & 15) + (((k32 >> 3) & 1) << 4)) * 20) +
         (k32 & 7) + (((k32 >> 4) & 1) << 3);
}
__device__ __forceinline__ int bSwz(int n, int k32) {
  return ((n >> 4) * 640) +
         (((n & 15) + (((k32 >> 3) & 1) << 4)) * 20) +
         (k32 & 7) + (((k32 >> 4) & 1) << 3);
}

__device__ __forceinline__ v16h ldsFragSwz(const _Float16* tile, int subtile) {
  union { v16h v; uint2 u[4]; } f;
  const uint2* p = reinterpret_cast<const uint2*>(
      tile + subtile * 640 + (threadIdx.x & 31) * 20);
#pragma unroll
  for (int j = 0; j < 4; ++j) f.u[j] = p[j];
  return f.v;
}

// ---------------------------------------------------------------------------
// Pass 0: fp32 -> fp16 conversion (streaming)
// ---------------------------------------------------------------------------
__global__ __launch_bounds__(256) void cvt_f32_f16(const float* __restrict__ s,
                                                   _Float16* __restrict__ d,
                                                   int n) {
  int i = blockIdx.x * blockDim.x + threadIdx.x;
  const int stride = gridDim.x * blockDim.x;
  for (; i < n; i += stride) d[i] = (_Float16)s[i];
}

// ---------------------------------------------------------------------------
// Pass 1: QKV GEMM [8192x1024]*[1024x3072]+bias.
// Block tile 256x128, 8 waves x (64x64), K-tile 32, double-buffered swizzled
// LDS, two-phase staging (all global loads in flight before LDS stores).
// Epilogue scatters Q,K head-major [B,H,T,D] and V transposed [B,H,D,T].
// ---------------------------------------------------------------------------
__global__ __launch_bounds__(256) void qkv_gemm(
    const _Float16* __restrict__ A, const _Float16* __restrict__ Bw,
    const float* __restrict__ bias, _Float16* __restrict__ Qf,
    _Float16* __restrict__ Kf, _Float16* __restrict__ Vt) {
  constexpr int K = 1024, N = 3072;
  __shared__ _Float16 As[2][16 * 640];  // 256 rows x 32 k, swizzled
  __shared__ _Float16 Bs[2][8 * 640];   // 128 cols x 32 k, swizzled
  const int tid = threadIdx.x;
  const int m0 = blockIdx.y * 256, n0 = blockIdx.x * 128;
  const int wid = tid >> 5;
  const int wm = (wid & 3) * 64, wn = (wid >> 2) * 64;

  auto stage = [&](int k0, int buf) {
    uint4 av[4], bv[2];
    // phase 1: issue all global loads (6 b128 per thread, all in flight)
#pragma unroll
    for (int c = 0; c < 4; ++c) {
      const int i = tid + c * 256;  // 1024 uint4's: 256 rows x 4
      const int row = i >> 2, kseg = (i & 3) * 8;
      av[c] = *reinterpret_cast<const uint4*>(A + (m0 + row) * K + k0 + kseg);
    }
#pragma unroll
    for (int c = 0; c < 2; ++c) {
      const int i = tid + c * 256;  // 512 uint4's: 32 k x 16 segs
      const int kk = i >> 4, nseg = (i & 15) * 8;
      bv[c] = *reinterpret_cast<const uint4*>(Bw + (k0 + kk) * N + n0 + nseg);
    }
    // phase 2: LDS stores (compiler staggers s_wait_loadcnt ladder)
#pragma unroll
    for (int c = 0; c < 4; ++c) {
      const int i = tid + c * 256;
      const int row = i >> 2, kseg = (i & 3) * 8;
      union { uint4 v; uint2 d[2]; } u; u.v = av[c];
      uint2* w = reinterpret_cast<uint2*>(&As[buf][aSwz(row, kseg)]);
      w[0] = u.d[0]; w[1] = u.d[1];
    }
#pragma unroll
    for (int c = 0; c < 2; ++c) {
      const int i = tid + c * 256;
      const int kk = i >> 4, nseg = (i & 15) * 8;
      union { uint4 v; _Float16 h[8]; } u; u.v = bv[c];
#pragma unroll
      for (int j = 0; j < 8; ++j) Bs[buf][bSwz(nseg + j, kk)] = u.h[j];
    }
  };

  v8f acc[4][4] = {};
  stage(0, 0);
  int buf = 0;
  for (int k0 = 0; k0 < K; k0 += 32) {
    __syncthreads();
    if (k0 + 32 < K) stage(k0 + 32, buf ^ 1);
    if (k0 + 64 < K) {  // L2 prefetch hint for the tile after next
      __builtin_prefetch(A + (m0 + tid) * K + k0 + 64, 0, 1);
      __builtin_prefetch(Bw + (k0 + 64 + (tid >> 4)) * N + n0 + (tid & 15) * 8,
                         0, 1);
    }
    v16h bf[4];
#pragma unroll
    for (int ni = 0; ni < 4; ++ni)
      bf[ni] = ldsFragSwz(Bs[buf], (wn >> 4) + ni);
#pragma unroll
    for (int mi = 0; mi < 4; ++mi) {
      v16h a = ldsFragSwz(As[buf], (wm >> 4) + mi);
#pragma unroll
      for (int ni = 0; ni < 4; ++ni)
        acc[mi][ni] = wmma32(a, bf[ni], acc[mi][ni]);
    }
    buf ^= 1;
  }

  // epilogue: bias + scatter into head-major Q/K and transposed V
  const int lane = tid & 31, r = lane & 15, hi = lane >> 4;
#pragma unroll
  for (int ni = 0; ni < 4; ++ni) {
    const int n = n0 + wn + ni * 16 + r;
    const float bv = bias[n];
    const int which = n >> 10, c = n & 1023, h = c >> 6, d = c & 63;
#pragma unroll
    for (int mi = 0; mi < 4; ++mi)
#pragma unroll
      for (int v = 0; v < 8; ++v) {
        const int m = m0 + wm + mi * 16 + v + 8 * hi;
        const int b = m >> 11, t = m & 2047;
        const _Float16 hv = (_Float16)(acc[mi][ni][v] + bv);
        const int bhh = b * 16 + h;
        if (which == 0)      Qf[(bhh * 2048 + t) * 64 + d] = hv;
        else if (which == 1) Kf[(bhh * 2048 + t) * 64 + d] = hv;
        else                 Vt[(bhh * 64 + d) * 2048 + t] = hv;
      }
  }
}

// ---------------------------------------------------------------------------
// Pass 2: causal flash attention. Block = 4 waves = 128 query rows of one
// (b,h); 64-wide key tiles, online softmax (wave32 shfl reductions), P written
// straight into swizzled A-fragment LDS layout from the D-layout registers.
// ---------------------------------------------------------------------------
__global__ __launch_bounds__(128) void attn_kernel(
    const _Float16* __restrict__ Qf, const _Float16* __restrict__ Kf,
    const _Float16* __restrict__ Vt, _Float16* __restrict__ AO) {
  __shared__ _Float16 Qs[2 * 5120];  // [kt][8 subtiles][640]  (128 x 64)
  __shared__ _Float16 Ks[2 * 2560];  // [kt][4 subtiles][640]  (64s x 64d)
  __shared__ _Float16 Vs[2 * 2560];  // [kt][4 subtiles][640]  (64d x 64s)
  __shared__ _Float16 Ps[4 * 2560];  // per-wave [kt][2 subtiles][640]

  const int tid = threadIdx.x, wid = tid >> 5;
  const int lane = tid & 31, r = lane & 15, hi = lane >> 4;
  const int t0 = blockIdx.x * 128;
  const int bh = blockIdx.y;
  const _Float16* qg = Qf + (bh * 2048 + t0) * 64;
  const _Float16* kgb = Kf + bh * 2048 * 64;
  const _Float16* vgb = Vt + bh * 64 * 2048;

  // stage Q tile [128][64] -> A-swizzled (two-phase)
  {
    uint4 qv[8];
#pragma unroll
    for (int c = 0; c < 8; ++c) {
      const int i = tid + c * 128;
      const int row = i >> 3, dseg = (i & 7) * 8;
      qv[c] = *reinterpret_cast<const uint4*>(qg + row * 64 + dseg);
    }
#pragma unroll
    for (int c = 0; c < 8; ++c) {
      const int i = tid + c * 128;
      const int row = i >> 3, dseg = (i & 7) * 8;
      union { uint4 v; uint2 d[2]; } u; u.v = qv[c];
      uint2* w = reinterpret_cast<uint2*>(
          &Qs[(dseg >> 5) * 5120 + aSwz(row, dseg & 31)]);
      w[0] = u.d[0]; w[1] = u.d[1];
    }
  }

  float mrow[2][8], lrow[2][8];
  v8f o[2][4] = {};
#pragma unroll
  for (int mi = 0; mi < 2; ++mi)
#pragma unroll
    for (int v = 0; v < 8; ++v) { mrow[mi][v] = -1e30f; lrow[mi][v] = 0.f; }

  const int qrow0 = wid * 32;
  const int nst = t0 / 64 + 2;
  for (int st = 0; st < nst; ++st) {
    const int s0 = st * 64;
    __syncthreads();
    // stage K [s][d] and V [d][s] -> B-swizzled (two-phase, 8 loads in flight)
    {
      uint4 kv[4], vv[4];
#pragma unroll
      for (int c = 0; c < 4; ++c) {
        const int i = tid + c * 128;
        const int row = i >> 3, seg = (i & 7) * 8;
        kv[c] = *reinterpret_cast<const uint4*>(kgb + (s0 + row) * 64 + seg);
        vv[c] = *reinterpret_cast<const uint4*>(vgb + row * 2048 + s0 + seg);
      }
#pragma unroll
      for (int c = 0; c < 4; ++c) {
        const int i = tid + c * 128;
        const int row = i >> 3, seg = (i & 7) * 8;
        union { uint4 v; uint2 d[2]; } uk, uv;
        uk.v = kv[c]; uv.v = vv[c];
        uint2* wk = reinterpret_cast<uint2*>(
            &Ks[(seg >> 5) * 2560 + bSwz(row, seg & 31)]);
        wk[0] = uk.d[0]; wk[1] = uk.d[1];
        uint2* wv = reinterpret_cast<uint2*>(
            &Vs[(seg >> 5) * 2560 + bSwz(row, seg & 31)]);
        wv[0] = uv.d[0]; wv[1] = uv.d[1];
      }
    }
    __syncthreads();

    // S = Q * K^T  (32x64 per wave)
    v8f sfr[2][4] = {};
#pragma unroll
    for (int mi = 0; mi < 2; ++mi)
#pragma unroll
      for (int kt = 0; kt < 2; ++kt) {
        v16h a = ldsFragSwz(Qs + kt * 5120, (qrow0 >> 4) + mi);
#pragma unroll
        for (int ni = 0; ni < 4; ++ni) {
          v16h b = ldsFragSwz(Ks + kt * 2560, ni);
          sfr[mi][ni] = wmma32(a, b, sfr[mi][ni]);
        }
      }

    // scale, causal mask, online softmax
#pragma unroll
    for (int mi = 0; mi < 2; ++mi) {
      float tm[8];
#pragma unroll
      for (int v = 0; v < 8; ++v) tm[v] = -1e30f;
#pragma unroll
      for (int ni = 0; ni < 4; ++ni)
#pragma unroll
        for (int v = 0; v < 8; ++v) {
          float x = sfr[mi][ni][v] * 0.125f;  // 1/sqrt(64)
          const int tq = t0 + qrow0 + mi * 16 + v + 8 * hi;
          const int sc = s0 + ni * 16 + r;
          if (sc > tq) x = -1e30f;
          sfr[mi][ni][v] = x;
          tm[v] = fmaxf(tm[v], x);
        }
#pragma unroll
      for (int v = 0; v < 8; ++v)
#pragma unroll
        for (int off = 1; off < 16; off <<= 1)
          tm[v] = fmaxf(tm[v], __shfl_xor(tm[v], off, 32));

      float mn[8], alpha[8], rs[8];
#pragma unroll
      for (int v = 0; v < 8; ++v) {
        mn[v] = fmaxf(mrow[mi][v], tm[v]);
        alpha[v] = __expf(mrow[mi][v] - mn[v]);
        mrow[mi][v] = mn[v];
        rs[v] = 0.f;
      }
#pragma unroll
      for (int ni = 0; ni < 4; ++ni)
#pragma unroll
        for (int v = 0; v < 8; ++v) {
          const float p = __expf(sfr[mi][ni][v] - mn[v]);
          sfr[mi][ni][v] = p;
          rs[v] += p;
        }
#pragma unroll
      for (int v = 0; v < 8; ++v) {
#pragma unroll
        for (int off = 1; off < 16; off <<= 1)
          rs[v] += __shfl_xor(rs[v], off, 32);
        lrow[mi][v] = lrow[mi][v] * alpha[v] + rs[v];
      }
#pragma unroll
      for (int ni = 0; ni < 4; ++ni)
#pragma unroll
        for (int v = 0; v < 8; ++v) o[mi][ni][v] *= alpha[v];
    }

    // P (D-layout f32) -> per-wave LDS, directly in A-swizzled f16 layout.
    // Per-wave region; LDS ops are in-order within a wave, no barrier needed.
    _Float16* myP = Ps + wid * 2560;
#pragma unroll
    for (int mi = 0; mi < 2; ++mi)
#pragma unroll
      for (int ni = 0; ni < 4; ++ni) {
        const int k = ni * 16 + r, kt = k >> 5, k32 = k & 31;
#pragma unroll
        for (int v = 0; v < 8; ++v)
          myP[kt * 1280 + mi * 640 + aSwz(v + 8 * hi, k32)] =
              (_Float16)sfr[mi][ni][v];
      }

    // O += P * V
#pragma unroll
    for (int mi = 0; mi < 2; ++mi)
#pragma unroll
      for (int kt = 0; kt < 2; ++kt) {
        v16h a = ldsFragSwz(myP + kt * 1280, mi);
#pragma unroll
        for (int ni = 0; ni < 4; ++ni) {
          v16h b = ldsFragSwz(Vs + kt * 2560, ni);
          o[mi][ni] = wmma32(a, b, o[mi][ni]);
        }
      }
  }

  // epilogue: normalize and store f16 [B,T,C]
  const int b = bh >> 4, h = bh & 15;
#pragma unroll
  for (int mi = 0; mi < 2; ++mi)
#pragma unroll
    for (int ni = 0; ni < 4; ++ni)
#pragma unroll
      for (int v = 0; v < 8; ++v) {
        const int tq = t0 + qrow0 + mi * 16 + v + 8 * hi;
        const float val = o[mi][ni][v] / lrow[mi][v];
        AO[(b * 2048 + tq) * 1024 + h * 64 + ni * 16 + r] = (_Float16)val;
      }
}

// ---------------------------------------------------------------------------
// Pass 3: projection GEMM [8192x1024]*[1024x1024]+bias -> f32 out.
// Same 256x128 / 64x64-wave double-buffered structure as pass 1.
// ---------------------------------------------------------------------------
__global__ __launch_bounds__(256) void proj_gemm(
    const _Float16* __restrict__ A, const _Float16* __restrict__ Bw,
    const float* __restrict__ bias, float* __restrict__ out) {
  constexpr int K = 1024, N = 1024;
  __shared__ _Float16 As[2][16 * 640];
  __shared__ _Float16 Bs[2][8 * 640];
  const int tid = threadIdx.x;
  const int m0 = blockIdx.y * 256, n0 = blockIdx.x * 128;
  const int wid = tid >> 5;
  const int wm = (wid & 3) * 64, wn = (wid >> 2) * 64;

  auto stage = [&](int k0, int buf) {
    uint4 av[4], bv[2];
#pragma unroll
    for (int c = 0; c < 4; ++c) {
      const int i = tid + c * 256;
      const int row = i >> 2, kseg = (i & 3) * 8;
      av[c] = *reinterpret_cast<const uint4*>(A + (m0 + row) * K + k0 + kseg);
    }
#pragma unroll
    for (int c = 0; c < 2; ++c) {
      const int i = tid + c * 256;
      const int kk = i >> 4, nseg = (i & 15) * 8;
      bv[c] = *reinterpret_cast<const uint4*>(Bw + (k0 + kk) * N + n0 + nseg);
    }
#pragma unroll
    for (int c = 0; c < 4; ++c) {
      const int i = tid + c * 256;
      const int row = i >> 2, kseg = (i & 3) * 8;
      union { uint4 v; uint2 d[2]; } u; u.v = av[c];
      uint2* w = reinterpret_cast<uint2*>(&As[buf][aSwz(row, kseg)]);
      w[0] = u.d[0]; w[1] = u.d[1];
    }
#pragma unroll
    for (int c = 0; c < 2; ++c) {
      const int i = tid + c * 256;
      const int kk = i >> 4, nseg = (i & 15) * 8;
      union { uint4 v; _Float16 h[8]; } u; u.v = bv[c];
#pragma unroll
      for (int j = 0; j < 8; ++j) Bs[buf][bSwz(nseg + j, kk)] = u.h[j];
    }
  };

  v8f acc[4][4] = {};
  stage(0, 0);
  int buf = 0;
  for (int k0 = 0; k0 < K; k0 += 32) {
    __syncthreads();
    if (k0 + 32 < K) stage(k0 + 32, buf ^ 1);
    if (k0 + 64 < K) {
      __builtin_prefetch(A + (m0 + tid) * K + k0 + 64, 0, 1);
      __builtin_prefetch(Bw + (k0 + 64 + (tid >> 4)) * N + n0 + (tid & 15) * 8,
                         0, 1);
    }
    v16h bf[4];
#pragma unroll
    for (int ni = 0; ni < 4; ++ni)
      bf[ni] = ldsFragSwz(Bs[buf], (wn >> 4) + ni);
#pragma unroll
    for (int mi = 0; mi < 4; ++mi) {
      v16h a = ldsFragSwz(As[buf], (wm >> 4) + mi);
#pragma unroll
      for (int ni = 0; ni < 4; ++ni)
        acc[mi][ni] = wmma32(a, bf[ni], acc[mi][ni]);
    }
    buf ^= 1;
  }

  const int lane = tid & 31, r = lane & 15, hi = lane >> 4;
#pragma unroll
  for (int ni = 0; ni < 4; ++ni) {
    const int n = n0 + wn + ni * 16 + r;
    const float bv = bias[n];
#pragma unroll
    for (int mi = 0; mi < 4; ++mi)
#pragma unroll
      for (int v = 0; v < 8; ++v) {
        const int m = m0 + wm + mi * 16 + v + 8 * hi;
        out[m * N + n] = acc[mi][ni][v] + bv;
      }
  }
}

// ---------------------------------------------------------------------------
// Host launcher
// ---------------------------------------------------------------------------
extern "C" void kernel_launch(void* const* d_in, const int* in_sizes, int n_in,
                              void* d_out, int out_size, void* d_ws,
                              size_t ws_size, hipStream_t stream) {
  (void)in_sizes; (void)n_in; (void)out_size; (void)ws_size;
  const float* x      = (const float*)d_in[0];  // [4,2048,1024]
  const float* W_attn = (const float*)d_in[1];  // [1024,3072]
  const float* b_attn = (const float*)d_in[2];  // [3072]
  const float* W_proj = (const float*)d_in[3];  // [1024,1024]
  const float* b_proj = (const float*)d_in[4];  // [1024]
  float* out = (float*)d_out;                   // [4,2048,1024]

  constexpr size_t NX = 8192ull * 1024;
  constexpr size_t NWA = 1024ull * 3072;
  constexpr size_t NWP = 1024ull * 1024;
  constexpr size_t NQKV = 4ull * 16 * 2048 * 64;

  _Float16* xh  = (_Float16*)d_ws;
  _Float16* wah = xh + NX;
  _Float16* wph = wah + NWA;
  _Float16* Qf  = wph + NWP;
  _Float16* Kf  = Qf + NQKV;
  _Float16* Vt  = Kf + NQKV;
  _Float16* AO  = Vt + NQKV;

  cvt_f32_f16<<<2048, 256, 0, stream>>>(x, xh, (int)NX);
  cvt_f32_f16<<<2048, 256, 0, stream>>>(W_attn, wah, (int)NWA);
  cvt_f32_f16<<<1024, 256, 0, stream>>>(W_proj, wph, (int)NWP);

  qkv_gemm<<<dim3(3072 / 128, 8192 / 256), 256, 0, stream>>>(
      xh, wah, b_attn, Qf, Kf, Vt);

  attn_kernel<<<dim3(2048 / 128, 4 * 16), 128, 0, stream>>>(Qf, Kf, Vt, AO);

  proj_gemm<<<dim3(1024 / 128, 8192 / 256), 256, 0, stream>>>(
      AO, wph, b_proj, out);
}
